// ShipGNNWithDecoder_25555055411313
// MI455X (gfx1250) — matrix operand order repaired and checked by reading, not compiled, not run on hardware
//
#include <hip/hip_runtime.h>
#include <hip/hip_bf16.h>

typedef __attribute__((ext_vector_type(16))) __bf16 v16bf;
typedef __attribute__((ext_vector_type(8)))  __bf16 v8bf;
typedef __attribute__((ext_vector_type(8)))  float  v8f;
typedef __attribute__((ext_vector_type(4)))  unsigned v4u;
typedef __attribute__((ext_vector_type(4)))  int      v4i;
typedef __attribute__((ext_vector_type(8)))  int      v8i;

#define Nn   100000
#define Ee   1600000
#define Hh   128
#define Gg   1000
#define GP   1008      // G padded to multiple of 16
#define NPGc 100
#define GAPI 4

static __device__ __forceinline__ __bf16 f2bf(float f) {
  union { float f; unsigned u; } v; v.f = f;
  unsigned r = v.u + 0x7FFFu + ((v.u >> 16) & 1u);   // round-to-nearest-even
  unsigned short h = (unsigned short)(r >> 16);
  return __builtin_bit_cast(__bf16, h);
}

// ---------------- degree / norm ----------------
__global__ __launch_bounds__(256) void k_fill1(float* p, int n) {
  int i = blockIdx.x * 256 + threadIdx.x;
  if (i < n) p[i] = 1.0f;
}

__global__ __launch_bounds__(256) void k_degcount(const int* __restrict__ dst,
                                                  float* __restrict__ deg, int e) {
  int i = blockIdx.x * 256 + threadIdx.x;
  if (i < e) atomicAdd(&deg[dst[i]], 1.0f);
}

__global__ __launch_bounds__(256) void k_rsqrt(const float* __restrict__ deg,
                                               float* __restrict__ dis, int n) {
  int i = blockIdx.x * 256 + threadIdx.x;
  if (i < n) dis[i] = rsqrtf(deg[i]);
}

// ---------------- feature build (bf16, K padded 18->32) ----------------
__global__ __launch_bounds__(256) void k_feats(const float* __restrict__ x,
                                               const float* __restrict__ se,
                                               const float* __restrict__ ne,
                                               __bf16* __restrict__ f, int n) {
  int i = blockIdx.x * 256 + threadIdx.x;
  if (i >= n) return;
  const float* xr = x + (size_t)i * 12;
  __bf16* fr = f + (size_t)i * 32;
#pragma unroll
  for (int c = 0; c < 10; ++c) fr[c] = f2bf(xr[c]);
  int st = (int)xr[10], nv = (int)xr[11];
#pragma unroll
  for (int c = 0; c < 4; ++c) fr[10 + c] = f2bf(se[st * 4 + c]);
#pragma unroll
  for (int c = 0; c < 4; ++c) fr[14 + c] = f2bf(ne[nv * 4 + c]);
#pragma unroll
  for (int c = 18; c < 32; ++c) fr[c] = f2bf(0.0f);
}

// ---------------- pack weight [Kreal x Ncols] f32 -> WMMA B fragments ----------------
// frag index = nt*KT + kt ; element = lane*16 + j ; B[k][n]: k = kt*32 + 8*(lane>>4) + (j&7) + 16*(j>>3), n = nt*16 + (lane&15)
__global__ __launch_bounds__(256) void k_packw(const float* __restrict__ W,
                                               __bf16* __restrict__ P,
                                               int Kreal, int KT, int NT, int Ncols) {
  int idx = blockIdx.x * 256 + threadIdx.x;
  int total = KT * NT * 512;
  if (idx >= total) return;
  int el = idx & 511, frag = idx >> 9;
  int kt = frag % KT, nt = frag / KT;
  int lane = el >> 4, j = el & 15;
  int k  = kt * 32 + (lane >> 4) * 8 + (j & 7) + 16 * (j >> 3);
  int nc = nt * 16 + (lane & 15);
  float v = (k < Kreal) ? W[(size_t)k * Ncols + nc] : 0.0f;
  P[idx] = f2bf(v);
}

// ---------------- WMMA GEMM: C[M,Ncols] = A[M,K](bf16) @ Bpacked + bias ----------------
// block = 8 waves; wave w -> 16x16 tile (rowTile = blockIdx.x, colTile = blockIdx.y*8+w).
// A tile (16 x K bf16, contiguous) is staged to LDS once per block by the Tensor Data
// Mover with hardware row padding (+16B/row) so per-lane ds_load_b128 fragment reads
// are bank-conflict-free. B fragments are contiguous 1KB chunks read from L2.
__global__ __launch_bounds__(256) void k_gemm(const __bf16* __restrict__ A,
                                              const __bf16* __restrict__ Bp,
                                              const float* __restrict__ bias,
                                              float* __restrict__ C,
                                              int K, int Ncols) {
  __shared__ __bf16 As[16 * 264];            // max K=256 -> row stride (K+8) halves
  const int lane = threadIdx.x & 31;
  const int wave = threadIdx.x >> 5;
  const int rowTile = blockIdx.x;
  const int colTile = blockIdx.y * 8 + wave;
  const int m  = lane & 15;
  const int hi = lane >> 4;
  const int KT = K >> 5;
  const int rs = K + 8;                      // padded LDS row stride (halves)

#if __has_builtin(__builtin_amdgcn_tensor_load_to_lds)
  if (threadIdx.x < 32) {                    // wave 0 issues the DMA (TDM ignores EXEC)
    unsigned ldsAddr = (unsigned)(size_t)(const void*)As;        // low 32b = LDS offset
    unsigned long long ga =
        (unsigned long long)(size_t)(const void*)(A + (size_t)rowTile * 16 * K);
    // elements are 4B words: row = K/2 words; pad_interval: 2^(v+1) dwords per row
    unsigned dim0 = (unsigned)(K >> 1);
    unsigned pint = (K == 32) ? 3u : (K == 128) ? 5u : 6u;
    // group0: count=1 | lds_addr | global_addr[56:0] | type=2
    v4u g0 = { 1u,
               ldsAddr,
               (unsigned)(ga & 0xffffffffu),
               (unsigned)((ga >> 32) & 0x01ffffffu) | (2u << 30) };
    // group1: data_size=4B(2), pad_enable, pad_interval, pad_amount=4dw(3);
    // tensor_dim0=dim0, tensor_dim1=16, tile_dim0=dim0, tile_dim1=16, tile_dim2=0,
    // tensor_dim0_stride=dim0 (contiguous rows), tensor_dim1_stride=0
    v8i g1 = { (int)((2u << 16) | (1u << 20) | (pint << 22) | (3u << 25)),
               (int)((dim0 & 0xffffu) << 16),
               (int)((dim0 >> 16) | (16u << 16)),
               (int)(dim0 << 16),
               (int)16,
               (int)dim0,
               0,
               0 };
    v4i gz = { 0, 0, 0, 0 };
#if defined(__clang_major__) && (__clang_major__ >= 23)
    v8i g2z = { 0, 0, 0, 0, 0, 0, 0, 0 };
    __builtin_amdgcn_tensor_load_to_lds(g0, g1, gz, gz, g2z, 0);
#else
    __builtin_amdgcn_tensor_load_to_lds(g0, g1, gz, gz, 0);
#endif
    __builtin_amdgcn_s_wait_tensorcnt(0);
  }
#else
  // fallback: cooperative staged copy into the same padded layout
  for (int c = threadIdx.x; c < 2 * K; c += 256) {     // 16 rows * (K/8) b128 chunks
    int row = c / (K >> 3), col = c % (K >> 3);
    *(v8bf*)(As + row * rs + col * 8) =
        *(const v8bf*)(A + ((size_t)rowTile * 16 + row) * K + col * 8);
  }
#endif
  __syncthreads();

  const __bf16* Al = As + m * rs + hi * 8;
  const __bf16* Bf = Bp + ((size_t)colTile * KT) * 512 + lane * 16;
  v8f acc = {};
  for (int kt = 0; kt < KT; ++kt) {
    // A 16x32 frag: lane m, halves j=0..7 -> K = kt*32 + 8*hi + j ; j=8..15 -> +16
    v8bf a0 = *(const v8bf*)(Al + kt * 32);
    v8bf a1 = *(const v8bf*)(Al + kt * 32 + 16);
    v8bf b0 = *(const v8bf*)(Bf + (size_t)kt * 512);
    v8bf b1 = *(const v8bf*)(Bf + (size_t)kt * 512 + 8);
    v16bf a, b;
#pragma unroll
    for (int i = 0; i < 8; ++i) { a[i] = a0[i]; a[i + 8] = a1[i]; b[i] = b0[i]; b[i + 8] = b1[i]; }
    acc = __builtin_amdgcn_wmma_f32_16x16x32_bf16(false, a, false, b, (short)0, acc, false, false);
  }
  const int cc = colTile * 16 + m;
  const float bv = bias ? bias[cc] : 0.0f;
  float* Cp = C + ((size_t)rowTile * 16 + 8 * hi) * Ncols + cc;
#pragma unroll
  for (int j = 0; j < 8; ++j) Cp[(size_t)j * Ncols] = acc[j] + bv;
}

// ---------------- GCN: out = m*self_norm + b (full init) ----------------
__global__ __launch_bounds__(256) void k_selfinit(const float* __restrict__ m,
                                                  const float* __restrict__ dis,
                                                  const float* __restrict__ b,
                                                  float* __restrict__ out, int n) {
  int i = blockIdx.x * 256 + threadIdx.x;
  if (i >= n) return;
  int node = i >> 7, c = i & 127;
  float d = dis[node];
  out[i] = m[i] * d * d + b[c];
}

// ---------------- GCN: edge scatter-add, 32 lanes/edge, float4/lane ----------------
__global__ __launch_bounds__(256) void k_edge(const float* __restrict__ m,
                                              const float* __restrict__ dis,
                                              const int* __restrict__ src,
                                              const int* __restrict__ dst,
                                              float* __restrict__ out, int e) {
  int t = blockIdx.x * 256 + threadIdx.x;
  int ei = t >> 5;
  if (ei >= e) return;
  int c4 = (t & 31) * 4;
  int s = src[ei], d = dst[ei];
  float nrm = dis[s] * dis[d];
  float4 mv = *(const float4*)(m + (size_t)s * Hh + c4);
  float* op = out + (size_t)d * Hh + c4;
  atomicAdd(op + 0, mv.x * nrm);
  atomicAdd(op + 1, mv.y * nrm);
  atomicAdd(op + 2, mv.z * nrm);
  atomicAdd(op + 3, mv.w * nrm);
}

// ---------------- f32 -> bf16 (optional relu) ----------------
__global__ __launch_bounds__(256) void k_cvtbf(const float* __restrict__ s,
                                               __bf16* __restrict__ d, int n, int relu) {
  int i = blockIdx.x * 256 + threadIdx.x;
  if (i >= n) return;
  float v = s[i];
  if (relu) v = fmaxf(v, 0.0f);
  d[i] = f2bf(v);
}

// ---------------- gather gap nodes -> GRU initial states (relu applied here) ----------------
__global__ __launch_bounds__(256) void k_gather(const float* __restrict__ h3,
                                                float* __restrict__ hff, float* __restrict__ hbf,
                                                __bf16* __restrict__ hfb, __bf16* __restrict__ hbb,
                                                __bf16* __restrict__ inpb) {
  int i = blockIdx.x * 256 + threadIdx.x;
  if (i >= GP * Hh) return;
  int g = i >> 7, c = i & 127;
  float v = 0.0f;
  if (g < Gg) v = fmaxf(h3[((size_t)g * NPGc + GAPI) * Hh + c], 0.0f);
  hff[i] = v; hbf[i] = v;
  __bf16 b = f2bf(v);
  hfb[i] = b; hbb[i] = b; inpb[i] = b;
}

// ---------------- GRU gate elementwise ----------------
__global__ __launch_bounds__(256) void k_gru(const float* __restrict__ gi,
                                             const float* __restrict__ gh,
                                             const float* __restrict__ h,
                                             float* __restrict__ hnf,
                                             __bf16* __restrict__ hnbf,
                                             __bf16* __restrict__ outcat, int half) {
  int i = blockIdx.x * 256 + threadIdx.x;
  if (i >= GP * Hh) return;
  int g = i >> 7, c = i & 127;
  const float* gir = gi + (size_t)g * 384;
  const float* ghr = gh + (size_t)g * 384;
  float r  = 1.0f / (1.0f + expf(-(gir[c] + ghr[c])));
  float z  = 1.0f / (1.0f + expf(-(gir[128 + c] + ghr[128 + c])));
  float nn = tanhf(gir[256 + c] + r * ghr[256 + c]);
  float hv = h[i];
  float hn = (1.0f - z) * nn + z * hv;
  hnf[i] = hn;
  hnbf[i] = f2bf(hn);
  outcat[(size_t)g * 256 + half * 128 + c] = f2bf(hn);
}

// ---------------- head: relu(t1) @ Wh2 + bh2 ----------------
__global__ __launch_bounds__(256) void k_pred(const float* __restrict__ t1,
                                              const float* __restrict__ Wh2,
                                              const float* __restrict__ bh2,
                                              float* __restrict__ out, int step) {
  int i = blockIdx.x * 256 + threadIdx.x;
  if (i >= Gg * 2) return;
  int g = i >> 1, c = i & 1;
  const float* r = t1 + (size_t)g * Hh;
  float s = bh2[c];
#pragma unroll 8
  for (int k = 0; k < Hh; ++k) s += fmaxf(r[k], 0.0f) * Wh2[k * 2 + c];
  out[(size_t)g * 6 + step * 2 + c] = s;
}

extern "C" void kernel_launch(void* const* d_in, const int* in_sizes, int n_in,
                              void* d_out, int out_size, void* d_ws, size_t ws_size,
                              hipStream_t stream) {
  const float* x     = (const float*)d_in[0];
  const int*   ei    = (const int*)  d_in[1];
  const float* ship  = (const float*)d_in[3];
  const float* nav   = (const float*)d_in[4];
  const float* W1    = (const float*)d_in[5];  const float* b1   = (const float*)d_in[6];
  const float* W2    = (const float*)d_in[7];  const float* b2   = (const float*)d_in[8];
  const float* W3    = (const float*)d_in[9];  const float* b3   = (const float*)d_in[10];
  const float* Wih_f = (const float*)d_in[11]; const float* Whh_f= (const float*)d_in[12];
  const float* bih_f = (const float*)d_in[13]; const float* bhh_f= (const float*)d_in[14];
  const float* Wih_b = (const float*)d_in[15]; const float* Whh_b= (const float*)d_in[16];
  const float* bih_b = (const float*)d_in[17]; const float* bhh_b= (const float*)d_in[18];
  const float* Wred  = (const float*)d_in[19]; const float* bred = (const float*)d_in[20];
  const float* Wh1   = (const float*)d_in[21]; const float* bh1  = (const float*)d_in[22];
  const float* Wh2   = (const float*)d_in[23]; const float* bh2  = (const float*)d_in[24];
  float* out = (float*)d_out;
  const int* src = ei;
  const int* dst = ei + Ee;

  char* wsp = (char*)d_ws;
  auto carve = [&](size_t bytes) {
    void* p = (void*)wsp;
    wsp += (bytes + 255) & ~(size_t)255;
    return p;
  };
  __bf16* featsb = (__bf16*)carve((size_t)Nn * 32 * 2);
  float*  deg    = (float*) carve((size_t)Nn * 4);
  float*  dis    = (float*) carve((size_t)Nn * 4);
  float*  mbuf   = (float*) carve((size_t)Nn * Hh * 4);
  float*  aggbuf = (float*) carve((size_t)Nn * Hh * 4);
  __bf16* nodebf = (__bf16*)carve((size_t)Nn * Hh * 2);
  __bf16* W1p    = (__bf16*)carve((size_t)1 * 8 * 512 * 2);
  __bf16* W2p    = (__bf16*)carve((size_t)4 * 8 * 512 * 2);
  __bf16* W3p    = (__bf16*)carve((size_t)4 * 8 * 512 * 2);
  __bf16* Wihfp  = (__bf16*)carve((size_t)4 * 24 * 512 * 2);
  __bf16* Whhfp  = (__bf16*)carve((size_t)4 * 24 * 512 * 2);
  __bf16* Wihbp  = (__bf16*)carve((size_t)4 * 24 * 512 * 2);
  __bf16* Whhbp  = (__bf16*)carve((size_t)4 * 24 * 512 * 2);
  __bf16* Wredp  = (__bf16*)carve((size_t)8 * 8 * 512 * 2);
  __bf16* Wh1p   = (__bf16*)carve((size_t)8 * 8 * 512 * 2);
  float*  gif    = (float*) carve((size_t)GP * 384 * 4);
  float*  ghf    = (float*) carve((size_t)GP * 384 * 4);
  float*  gib    = (float*) carve((size_t)GP * 384 * 4);
  float*  ghb    = (float*) carve((size_t)GP * 384 * 4);
  float*  hf_f   = (float*) carve((size_t)GP * Hh * 4);
  float*  hb_f   = (float*) carve((size_t)GP * Hh * 4);
  __bf16* hf_b   = (__bf16*)carve((size_t)GP * Hh * 2);
  __bf16* hb_b   = (__bf16*)carve((size_t)GP * Hh * 2);
  __bf16* inp_b  = (__bf16*)carve((size_t)GP * Hh * 2);
  __bf16* cat_b  = (__bf16*)carve((size_t)GP * 256 * 2);
  float*  t1f    = (float*) carve((size_t)GP * Hh * 4);
  float*  inpf   = (float*) carve((size_t)GP * Hh * 4);

  // degrees + symmetric norm
  k_fill1   <<<(Nn + 255) / 256, 256, 0, stream>>>(deg, Nn);
  k_degcount<<<(Ee + 255) / 256, 256, 0, stream>>>(dst, deg, Ee);
  k_rsqrt   <<<(Nn + 255) / 256, 256, 0, stream>>>(deg, dis, Nn);

  // features + weight packing
  k_feats<<<(Nn + 255) / 256, 256, 0, stream>>>(x, ship, nav, featsb, Nn);
  auto pack = [&](const float* W, __bf16* P, int Kreal, int Kpad, int Ncols) {
    int KT = Kpad / 32, NT = Ncols / 16, tot = KT * NT * 512;
    k_packw<<<(tot + 255) / 256, 256, 0, stream>>>(W, P, Kreal, KT, NT, Ncols);
  };
  pack(W1, W1p, 18, 32, 128);
  pack(W2, W2p, 128, 128, 128);
  pack(W3, W3p, 128, 128, 128);
  pack(Wih_f, Wihfp, 128, 128, 384);
  pack(Whh_f, Whhfp, 128, 128, 384);
  pack(Wih_b, Wihbp, 128, 128, 384);
  pack(Whh_b, Whhbp, 128, 128, 384);
  pack(Wred, Wredp, 256, 256, 128);
  pack(Wh1,  Wh1p,  256, 256, 128);

  // 3 GCN layers
  const float* bs[3] = {b1, b2, b3};
  for (int L = 0; L < 3; ++L) {
    if (L == 0)
      k_gemm<<<dim3(Nn / 16, 1), 256, 0, stream>>>(featsb, W1p, nullptr, mbuf, 32, 128);
    else
      k_gemm<<<dim3(Nn / 16, 1), 256, 0, stream>>>(nodebf, (L == 1 ? W2p : W3p), nullptr, mbuf, 128, 128);
    k_selfinit<<<(Nn * Hh + 255) / 256, 256, 0, stream>>>(mbuf, dis, bs[L], aggbuf, Nn * Hh);
    k_edge<<<(Ee * 32 + 255) / 256, 256, 0, stream>>>(mbuf, dis, src, dst, aggbuf, Ee);
    if (L < 2)
      k_cvtbf<<<(Nn * Hh + 255) / 256, 256, 0, stream>>>(aggbuf, nodebf, Nn * Hh, 1);
  }

  // gather gap nodes (relu of layer-3 output)
  k_gather<<<(GP * Hh + 255) / 256, 256, 0, stream>>>(aggbuf, hf_f, hb_f, hf_b, hb_b, inp_b);

  // GRU decoder, 3 steps
  for (int t = 0; t < 3; ++t) {
    k_gemm<<<dim3(GP / 16, 3), 256, 0, stream>>>(inp_b, Wihfp, bih_f, gif, 128, 384);
    k_gemm<<<dim3(GP / 16, 3), 256, 0, stream>>>(hf_b,  Whhfp, bhh_f, ghf, 128, 384);
    k_gemm<<<dim3(GP / 16, 3), 256, 0, stream>>>(inp_b, Wihbp, bih_b, gib, 128, 384);
    k_gemm<<<dim3(GP / 16, 3), 256, 0, stream>>>(hb_b,  Whhbp, bhh_b, ghb, 128, 384);
    k_gru<<<(GP * Hh + 255) / 256, 256, 0, stream>>>(gif, ghf, hf_f, hf_f, hf_b, cat_b, 0);
    k_gru<<<(GP * Hh + 255) / 256, 256, 0, stream>>>(gib, ghb, hb_f, hb_f, hb_b, cat_b, 1);
    k_gemm<<<dim3(GP / 16, 1), 256, 0, stream>>>(cat_b, Wh1p, bh1, t1f, 256, 128);
    k_pred<<<(Gg * 2 + 255) / 256, 256, 0, stream>>>(t1f, Wh2, bh2, out, t);
    k_gemm<<<dim3(GP / 16, 1), 256, 0, stream>>>(cat_b, Wredp, bred, inpf, 256, 128);
    k_cvtbf<<<(GP * Hh + 255) / 256, 256, 0, stream>>>(inpf, inp_b, GP * Hh, 0);
  }
}
